// MoELayer_61589831024933
// MI455X (gfx1250) — matrix operand order repaired and checked by reading, not compiled, not run on hardware
//
#include <hip/hip_runtime.h>
#include <hip/hip_bf16.h>
#include <math.h>

// ---------------- types ----------------
typedef __bf16 bf16_t;
typedef bf16_t v16bf __attribute__((ext_vector_type(16)));
typedef bf16_t v8bf  __attribute__((ext_vector_type(8)));
typedef bf16_t v4bf  __attribute__((ext_vector_type(4)));
typedef float  v8f   __attribute__((ext_vector_type(8)));

// 16-byte int vector matching the async-copy builtin's pointee type
typedef int v4i32 __attribute__((vector_size(16)));
typedef __attribute__((address_space(1))) v4i32 g_v4i32;   // global
typedef __attribute__((address_space(3))) v4i32 l_v4i32;   // LDS

#define D_DIM 1024
#define E_NUM 8
#define F_DIM 4096

// LDS tile geometry: K-step 32, rows padded to 40 elems (80B) for bank spread.
#define TSTR 40

// ---------------- CDNA5 async copy (guarded) ----------------
#if defined(__gfx1250__) && __has_builtin(__builtin_amdgcn_global_load_async_to_lds_b128)
#define HAVE_ASYNC 1
#else
#define HAVE_ASYNC 0
#endif

__device__ __forceinline__ void cp16_g2l(const bf16_t* g, bf16_t* l) {
#if HAVE_ASYNC
    __builtin_amdgcn_global_load_async_to_lds_b128(
        (g_v4i32*)g, (l_v4i32*)l, 0, 0);
#else
    *(v8bf*)l = *(const v8bf*)g;
#endif
}

__device__ __forceinline__ void wait_async_copies() {
#if HAVE_ASYNC
#if __has_builtin(__builtin_amdgcn_s_wait_asynccnt)
    __builtin_amdgcn_s_wait_asynccnt(0);
#else
    asm volatile("s_wait_asynccnt 0x0" ::: "memory");
#endif
#endif
}

// ---------------- tiny init ----------------
__global__ void moe_init_cnt(int* cnt) {
    if (threadIdx.x < E_NUM) cnt[threadIdx.x] = 0;
}

// ---------------- x fp32 -> bf16 ----------------
__global__ void moe_cvt_x(const float* __restrict__ x, bf16_t* __restrict__ xb, int n4) {
    int i = blockIdx.x * blockDim.x + threadIdx.x;
    if (i >= n4) return;
    float4 v = ((const float4*)x)[i];
    v4bf o = { (bf16_t)v.x, (bf16_t)v.y, (bf16_t)v.z, (bf16_t)v.w };
    *(v4bf*)(xb + (size_t)i * 4) = o;
}

// ---------------- transpose + convert: in[z][R][C] f32 -> out[z][C][R] bf16 ----------------
__global__ void moe_transpose_cvt(const float* __restrict__ in, bf16_t* __restrict__ out,
                                  int R, int C) {
    __shared__ float tile[32][33];
    const size_t zofs = (size_t)blockIdx.z * R * C;
    in  += zofs;
    out += zofs;
    const int tx = threadIdx.x, ty = threadIdx.y;
    const int r0 = blockIdx.y * 32, c0 = blockIdx.x * 32;
#pragma unroll
    for (int i = 0; i < 4; ++i)
        tile[ty + i * 8][tx] = in[(size_t)(r0 + ty + i * 8) * C + c0 + tx];
    __syncthreads();
#pragma unroll
    for (int i = 0; i < 4; ++i)
        out[(size_t)(c0 + ty + i * 8) * R + r0 + tx] = (bf16_t)tile[tx][ty + i * 8];
}

// ---------------- router: one wave (32 lanes) per token ----------------
__global__ void moe_router(const float* __restrict__ x, const float* __restrict__ gw,
                           const float* __restrict__ gb, float* __restrict__ wt,
                           int* __restrict__ list, int* __restrict__ cnt, int T) {
    const int wave = threadIdx.x >> 5;
    const int lane = threadIdx.x & 31;
    const int t = blockIdx.x * 8 + wave;
    if (t >= T) return;

    float acc[E_NUM];
#pragma unroll
    for (int e = 0; e < E_NUM; ++e) acc[e] = 0.0f;

    const float* xr = x + (size_t)t * D_DIM;
    for (int d = lane; d < D_DIM; d += 32) {
        float xv = xr[d];
#pragma unroll
        for (int e = 0; e < E_NUM; ++e) acc[e] += xv * gw[d * E_NUM + e];
    }
#pragma unroll
    for (int e = 0; e < E_NUM; ++e) {
#pragma unroll
        for (int off = 16; off >= 1; off >>= 1)
            acc[e] += __shfl_xor(acc[e], off, 32);
    }
    if (lane == 0) {
        float lg[E_NUM];
#pragma unroll
        for (int e = 0; e < E_NUM; ++e) lg[e] = acc[e] + gb[e];
        int i0 = 0;
#pragma unroll
        for (int e = 1; e < E_NUM; ++e) if (lg[e] > lg[i0]) i0 = e;
        int i1 = (i0 == 0) ? 1 : 0;
#pragma unroll
        for (int e = 0; e < E_NUM; ++e)
            if (e != i0 && lg[e] > lg[i1]) i1 = e;
        float eb = __expf(lg[i1] - lg[i0]);
        float inv = 1.0f / (1.0f + eb);
        wt[t * 2 + 0] = inv;
        wt[t * 2 + 1] = eb * inv;
        int p0 = atomicAdd(&cnt[i0], 1);
        list[i0 * T + p0] = t * 2 + 0;
        int p1 = atomicAdd(&cnt[i1], 1);
        list[i1 * T + p1] = t * 2 + 1;
    }
}

// ---------------- expert row offsets (prefix sum over 8) ----------------
__global__ void moe_offsets(const int* __restrict__ cnt, int* __restrict__ off) {
    if (threadIdx.x == 0) {
        int s = 0;
        for (int e = 0; e < E_NUM; ++e) { off[e] = s; s += cnt[e]; }
    }
}

// ---------------- helpers ----------------
__device__ __forceinline__ v16bf mk_frag(v8bf lo, v8bf hi) {
    return __builtin_shufflevector(lo, hi, 0, 1, 2, 3, 4, 5, 6, 7,
                                   8, 9, 10, 11, 12, 13, 14, 15);
}

// ==========================================================================
// pass A: H = gelu(gather(X) @ W1 + b1), bf16 out
// block = 256 thr = 8 waves; block tile M=64 x N=256; wave tile 32x64; K-step 32.
// A and B tiles double-buffered in LDS via async global->LDS copies.
// ==========================================================================
__global__ __launch_bounds__(256) void moe_ffn1(
    const bf16_t* __restrict__ xb, const bf16_t* __restrict__ w1t,
    const float* __restrict__ b1, const int* __restrict__ list,
    const int* __restrict__ cnt, const int* __restrict__ off,
    bf16_t* __restrict__ H, int T) {
    __shared__ bf16_t As[2][64][TSTR];
    __shared__ bf16_t Bs[2][256][TSTR];
    __shared__ unsigned agofs[64];

    const int e = blockIdx.z;
    const int cntE = cnt[e];
    const int m_base = blockIdx.y * 64;
    if (m_base >= cntE) return;

    const int tid = threadIdx.x;
    // token row -> element offset in xb
    if (tid < 64) {
        int row = m_base + tid;
        int rc = row < cntE ? row : cntE - 1;
        agofs[tid] = (unsigned)((list[e * T + rc] >> 1) * D_DIM);
    }
    __syncthreads();

    const int arow = tid >> 2, ach = (tid & 3) * 8;     // A copy: 1x16B per thread
    const unsigned aofs = agofs[arow];
    const unsigned bcol0 = (unsigned)e * F_DIM + blockIdx.x * 256;

    auto stage = [&](int buf, int k) {
        cp16_g2l(xb + aofs + k + ach, &As[buf][arow][ach]);
#pragma unroll
        for (int i = 0; i < 4; ++i) {                   // B copy: 4x16B per thread
            int idx = tid + i * 256;
            int col = idx >> 2, ch = (idx & 3) * 8;
            cp16_g2l(w1t + (size_t)(bcol0 + col) * D_DIM + k + ch, &Bs[buf][col][ch]);
        }
    };

    const int lane = tid & 31, wave = tid >> 5;
    const int wm = wave & 1, wn = wave >> 1;
    const int lr = lane & 15, kh = lane >> 4;
    const int n_base = blockIdx.x * 256 + wn * 64;

    v8f acc[2][4];
#pragma unroll
    for (int i = 0; i < 2; ++i)
#pragma unroll
        for (int j = 0; j < 4; ++j) acc[i][j] = (v8f)(0.0f);

    stage(0, 0);
    wait_async_copies();
    __syncthreads();

    const int nsteps = D_DIM / 32;
    for (int ks = 0; ks < nsteps; ++ks) {
        const int buf = ks & 1;
        if (ks + 1 < nsteps) stage(buf ^ 1, (ks + 1) * 32);

        v16bf a[2], b[4];
#pragma unroll
        for (int i = 0; i < 2; ++i) {
            const bf16_t* p = &As[buf][wm * 32 + i * 16 + lr][kh * 8];
            a[i] = mk_frag(*(const v8bf*)p, *(const v8bf*)(p + 16));
        }
#pragma unroll
        for (int j = 0; j < 4; ++j) {
            const bf16_t* p = &Bs[buf][wn * 64 + j * 16 + lr][kh * 8];
            b[j] = mk_frag(*(const v8bf*)p, *(const v8bf*)(p + 16));
        }
#pragma unroll
        for (int i = 0; i < 2; ++i)
#pragma unroll
            for (int j = 0; j < 4; ++j)
                acc[i][j] = __builtin_amdgcn_wmma_f32_16x16x32_bf16(
                    false, a[i], false, b[j], (short)0, acc[i][j], false, false);

        wait_async_copies();
        __syncthreads();
    }

    const size_t hbase = (size_t)off[e];
#pragma unroll
    for (int i = 0; i < 2; ++i) {
#pragma unroll
        for (int j = 0; j < 4; ++j) {
            const int col = n_base + j * 16 + lr;
            const float bias = b1[e * F_DIM + col];
#pragma unroll
            for (int r = 0; r < 8; ++r) {
                int g = m_base + wm * 32 + i * 16 + kh * 8 + r;
                if (g < cntE) {
                    float v = acc[i][j][r] + bias;
                    v = 0.5f * v * (1.0f + erff(v * 0.70710678118654752f));
                    H[(hbase + (size_t)g) * F_DIM + col] = (bf16_t)v;
                }
            }
        }
    }
}

// ==========================================================================
// pass B: slot[(k,t)] = H @ W2 + b2, f32 out. Same tiling, K = 4096.
// ==========================================================================
__global__ __launch_bounds__(256) void moe_ffn2(
    const bf16_t* __restrict__ H, const bf16_t* __restrict__ w2t,
    const float* __restrict__ b2, const int* __restrict__ list,
    const int* __restrict__ cnt, const int* __restrict__ off,
    float* __restrict__ slot, int T) {
    __shared__ bf16_t As[2][64][TSTR];
    __shared__ bf16_t Bs[2][256][TSTR];
    __shared__ unsigned agofs[64];

    const int e = blockIdx.z;
    const int cntE = cnt[e];
    const int m_base = blockIdx.y * 64;
    if (m_base >= cntE) return;

    const size_t hbase = (size_t)off[e];
    const int tid = threadIdx.x;
    if (tid < 64) {
        int row = m_base + tid;
        int rc = row < cntE ? row : cntE - 1;
        agofs[tid] = (unsigned)((hbase + (size_t)rc) * F_DIM);
    }
    __syncthreads();

    const int arow = tid >> 2, ach = (tid & 3) * 8;
    const unsigned aofs = agofs[arow];
    const unsigned bcol0 = (unsigned)e * D_DIM + blockIdx.x * 256;

    auto stage = [&](int buf, int k) {
        cp16_g2l(H + aofs + k + ach, &As[buf][arow][ach]);
#pragma unroll
        for (int i = 0; i < 4; ++i) {
            int idx = tid + i * 256;
            int col = idx >> 2, ch = (idx & 3) * 8;
            cp16_g2l(w2t + (size_t)(bcol0 + col) * F_DIM + k + ch, &Bs[buf][col][ch]);
        }
    };

    const int lane = tid & 31, wave = tid >> 5;
    const int wm = wave & 1, wn = wave >> 1;
    const int lr = lane & 15, kh = lane >> 4;
    const int n_base = blockIdx.x * 256 + wn * 64;

    v8f acc[2][4];
#pragma unroll
    for (int i = 0; i < 2; ++i)
#pragma unroll
        for (int j = 0; j < 4; ++j) acc[i][j] = (v8f)(0.0f);

    stage(0, 0);
    wait_async_copies();
    __syncthreads();

    const int nsteps = F_DIM / 32;
    for (int ks = 0; ks < nsteps; ++ks) {
        const int buf = ks & 1;
        if (ks + 1 < nsteps) stage(buf ^ 1, (ks + 1) * 32);

        v16bf a[2], b[4];
#pragma unroll
        for (int i = 0; i < 2; ++i) {
            const bf16_t* p = &As[buf][wm * 32 + i * 16 + lr][kh * 8];
            a[i] = mk_frag(*(const v8bf*)p, *(const v8bf*)(p + 16));
        }
#pragma unroll
        for (int j = 0; j < 4; ++j) {
            const bf16_t* p = &Bs[buf][wn * 64 + j * 16 + lr][kh * 8];
            b[j] = mk_frag(*(const v8bf*)p, *(const v8bf*)(p + 16));
        }
#pragma unroll
        for (int i = 0; i < 2; ++i)
#pragma unroll
            for (int j = 0; j < 4; ++j)
                acc[i][j] = __builtin_amdgcn_wmma_f32_16x16x32_bf16(
                    false, a[i], false, b[j], (short)0, acc[i][j], false, false);

        wait_async_copies();
        __syncthreads();
    }

#pragma unroll
    for (int i = 0; i < 2; ++i) {
#pragma unroll
        for (int j = 0; j < 4; ++j) {
            const int col = n_base + j * 16 + lr;
            const float bias = b2[e * D_DIM + col];
#pragma unroll
            for (int r = 0; r < 8; ++r) {
                int g = m_base + wm * 32 + i * 16 + kh * 8 + r;
                if (g < cntE) {
                    int enc = list[e * T + g];
                    int t = enc >> 1, kk = enc & 1;
                    slot[((size_t)kk * T + t) * D_DIM + col] = acc[i][j][r] + bias;
                }
            }
        }
    }
}

// ---------------- combine: out = w0*slot0 + w1*slot1 ----------------
__global__ void moe_combine(const float* __restrict__ slot, const float* __restrict__ wt,
                            float* __restrict__ out, int T) {
    int i = blockIdx.x * blockDim.x + threadIdx.x;          // float4 index
    int n4 = T * D_DIM / 4;
    if (i >= n4) return;
    int t = (i * 4) / D_DIM;
    float w0 = wt[t * 2 + 0], w1 = wt[t * 2 + 1];
    const float4 a = ((const float4*)slot)[i];
    const float4 b = ((const float4*)(slot + (size_t)T * D_DIM))[i];
    float4 o;
    o.x = w0 * a.x + w1 * b.x;
    o.y = w0 * a.y + w1 * b.y;
    o.z = w0 * a.z + w1 * b.z;
    o.w = w0 * a.w + w1 * b.w;
    ((float4*)out)[i] = o;
}

// ---------------- launch ----------------
extern "C" void kernel_launch(void* const* d_in, const int* in_sizes, int n_in,
                              void* d_out, int out_size, void* d_ws, size_t ws_size,
                              hipStream_t stream) {
    const float* x   = (const float*)d_in[0];
    const float* gw  = (const float*)d_in[1];
    const float* gb  = (const float*)d_in[2];
    const float* w1  = (const float*)d_in[3];
    const float* b1  = (const float*)d_in[4];
    const float* w2  = (const float*)d_in[5];
    const float* b2  = (const float*)d_in[6];
    float* out = (float*)d_out;

    const int T = in_sizes[0] / D_DIM;   // 8192

    // workspace carve-up
    char* p = (char*)d_ws;
    auto carve = [&](size_t bytes) {
        void* r = (void*)p;
        p += (bytes + 255) & ~(size_t)255;
        return r;
    };
    bf16_t* xb   = (bf16_t*)carve((size_t)T * D_DIM * 2);
    bf16_t* w1t  = (bf16_t*)carve((size_t)E_NUM * D_DIM * F_DIM * 2);
    bf16_t* w2t  = (bf16_t*)carve((size_t)E_NUM * D_DIM * F_DIM * 2);
    bf16_t* H    = (bf16_t*)carve((size_t)2 * T * F_DIM * 2);
    float*  slot = (float*) carve((size_t)2 * T * D_DIM * 4);
    float*  wt   = (float*) carve((size_t)T * 2 * 4);
    int*    list = (int*)   carve((size_t)E_NUM * T * 4);
    int*    cnt  = (int*)   carve(E_NUM * 4);
    int*    off  = (int*)   carve(E_NUM * 4);
    (void)ws_size; (void)n_in; (void)out_size;

    moe_init_cnt<<<1, 32, 0, stream>>>(cnt);

    {
        int n4 = T * D_DIM / 4;
        moe_cvt_x<<<(n4 + 255) / 256, 256, 0, stream>>>(x, xb, n4);
    }
    {   // w1 [E][D][F] -> w1t [E][F][D]
        dim3 g(F_DIM / 32, D_DIM / 32, E_NUM);
        moe_transpose_cvt<<<g, dim3(32, 8, 1), 0, stream>>>(w1, w1t, D_DIM, F_DIM);
    }
    {   // w2 [E][F][D] -> w2t [E][D][F]
        dim3 g(D_DIM / 32, F_DIM / 32, E_NUM);
        moe_transpose_cvt<<<g, dim3(32, 8, 1), 0, stream>>>(w2, w2t, F_DIM, D_DIM);
    }

    moe_router<<<(T + 7) / 8, 256, 0, stream>>>(x, gw, gb, wt, list, cnt, T);
    moe_offsets<<<1, 32, 0, stream>>>(cnt, off);

    {
        dim3 g(F_DIM / 256, (T + 63) / 64, E_NUM);
        moe_ffn1<<<g, 256, 0, stream>>>(xb, w1t, b1, list, cnt, off, H, T);
    }
    {
        dim3 g(D_DIM / 256, (T + 63) / 64, E_NUM);
        moe_ffn2<<<g, 256, 0, stream>>>(H, w2t, b2, list, cnt, off, slot, T);
    }
    {
        int n4 = T * D_DIM / 4;
        moe_combine<<<(n4 + 255) / 256, 256, 0, stream>>>(slot, wt, out, T);
    }
}
